// CausalSelfAttention_65146063946528
// MI455X (gfx1250) — compile-verified, matrix operand order checked by reading, not run on hardware
//
#include <hip/hip_runtime.h>

typedef __attribute__((ext_vector_type(16))) _Float16 v16h;
typedef __attribute__((ext_vector_type(8)))  _Float16 v8h;
typedef __attribute__((ext_vector_type(8)))  float    v8f;
typedef __attribute__((ext_vector_type(4)))  unsigned int v4u;
typedef __attribute__((ext_vector_type(8)))  int      v8i;
typedef __attribute__((ext_vector_type(4)))  int      v4i;

#define B_  2
#define T_  2048
#define D_  1024
#define H_  16
#define HD_ 64

// D = A(16x32 f16) x B(32x16 f16) + C(16x16 f32)
__device__ __forceinline__ v8f wmma_f16(v16h a, v16h b, v8f c) {
    return __builtin_amdgcn_wmma_f32_16x16x32_f16(false, a, false, b,
                                                  (short)0, c, false, false);
}

// Load one 16x32 A-fragment (or 32x16 B-fragment from column-major storage).
// Per CDNA5 ISA 7.12.2: lanes 0-15 hold row r, K = {0..7, 16..23};
// lanes 16-31 hold row r, K = {8..15, 24..31}.  Two 16B loads + shuffle.
__device__ __forceinline__ v16h load_frag(const _Float16* base, int ld) {
    int lane = threadIdx.x & 31;
    int r    = lane & 15;
    int ko   = (lane >> 4) << 3;      // 0 or 8
    const _Float16* p = base + (size_t)r * ld + ko;
    v8h lo = *(const v8h*)(p);        // K = ko .. ko+7
    v8h hi = *(const v8h*)(p + 16);   // K = ko+16 .. ko+23
    return __builtin_shufflevector(lo, hi, 0,1,2,3,4,5,6,7,8,9,10,11,12,13,14,15);
}

// ------------------------------------------------------------------- TDM ----
#if __has_builtin(__builtin_amdgcn_tensor_load_to_lds)
#define HAVE_TDM 1
#else
#define HAVE_TDM 0
#endif

__device__ __forceinline__ unsigned lds_off(const void* p) {
    // LDS aperture: addr[31:0] is the in-allocation byte offset.
    return (unsigned)(size_t)p;
}

#if HAVE_TDM
// 2D f16 tile load, Memory -> LDS, via Tensor Data Mover.
// tile_d0 = contiguous elements per row, tile_d1 = rows,
// stride_e = row stride in elements.  LDS result: [tile_d1][tile_d0] packed.
__device__ __forceinline__ void tdm_load_2d(unsigned lds_addr, const void* gaddr,
                                            unsigned tile_d0, unsigned tile_d1,
                                            unsigned long long stride_e) {
    unsigned long long ga = (unsigned long long)(size_t)gaddr;
    v4u g0;
    g0[0] = 1u;                                        // count=1, user D#
    g0[1] = lds_addr;                                  // lds_addr
    g0[2] = (unsigned)(ga & 0xFFFFFFFFu);              // global_addr[31:0]
    g0[3] = (unsigned)((ga >> 32) & 0x01FFFFFFu)       // global_addr[56:32]
          | (2u << 30);                                // type=2 ("image")
    v8i g1;
    g1[0] = (int)(1u << 16);                           // data_size=1 (2 bytes)
    g1[1] = (int)((tile_d0 & 0xFFFFu) << 16);          // tensor_dim0[15:0] @63:48
    g1[2] = (int)((tile_d1 & 0xFFFFu) << 16);          // tensor_dim1[15:0] @95:80
    g1[3] = (int)((tile_d0 & 0xFFFFu) << 16);          // tile_dim0 @127:112
    g1[4] = (int)(tile_d1 & 0xFFFFu);                  // tile_dim1 @143:128, tile_dim2=0
    g1[5] = (int)(unsigned)(stride_e & 0xFFFFFFFFu);   // dim0_stride[31:0]
    g1[6] = (int)(unsigned)((stride_e >> 32) & 0xFFFFu); // dim0_stride[47:32]
    g1[7] = 0;                                         // dim1_stride hi (unused, 2D)
    v4i z4 = {0, 0, 0, 0};
#if __clang_major__ >= 23
    v8i z8 = {0, 0, 0, 0, 0, 0, 0, 0};
    __builtin_amdgcn_tensor_load_to_lds(g0, g1, z4, z4, z8, 0);
#else
    __builtin_amdgcn_tensor_load_to_lds(g0, g1, z4, z4, 0);
#endif
}
#endif

// ---------------------------------------------------------------- converts --
__global__ void cvt_f32_to_f16(const float* __restrict__ in,
                               _Float16* __restrict__ out, int n) {
    for (int i = blockIdx.x * blockDim.x + threadIdx.x; i < n;
         i += gridDim.x * blockDim.x)
        out[i] = (_Float16)in[i];
}

// out[n*K + k] = (f16) in[k*N + n]   (LDS-tiled 32x32 transpose, f32 -> f16)
__global__ void transpose_cvt(const float* __restrict__ in,
                              _Float16* __restrict__ out, int K, int N) {
    __shared__ float tile[32][33];
    int n0 = blockIdx.x * 32;
    int k0 = blockIdx.y * 32;
    int tx = threadIdx.x, ty = threadIdx.y;       // 32 x 8
    for (int i = ty; i < 32; i += 8)
        tile[i][tx] = in[(size_t)(k0 + i) * N + (n0 + tx)];
    __syncthreads();
    for (int i = ty; i < 32; i += 8)
        out[(size_t)(n0 + i) * K + (k0 + tx)] = (_Float16)tile[tx][i];
}

// ---------------------------------------------------------------- QKV GEMM --
// qkv[4096,3072] = xh[4096,1024] @ WqkvT'(col-major f16) + bqkv
// One wave per 32x64 macro-tile: 2 A-frags x 4 B-frags -> 8 WMMAs / 32-K step.
// acc = 8 x v8f = 64 VGPRs; 6 live fragments = 48 VGPRs -> no spilling.
// Scatter into Q/K (row-major [BH,T,HD]) and V transposed ([BH,HD,T]).
__global__ void __launch_bounds__(256, 1)
gemm_qkv(const _Float16* __restrict__ xh,
         const _Float16* __restrict__ WT,
         const float* __restrict__ bias,
         _Float16* __restrict__ Q,
         _Float16* __restrict__ Kb,
         _Float16* __restrict__ Vt) {
    int wg = (blockIdx.x * blockDim.x + threadIdx.x) >> 5;
    const int TN = (3 * D_) / 64;                 // 48 macro-tiles in N
    int tm = wg / TN, tn = wg % TN;               // tm: 32-row, tn: 64-col
    const _Float16* A0 = xh + (size_t)tm * 32 * D_;
    const _Float16* B0 = WT + (size_t)tn * 64 * D_;
    v8f acc[2][4] = {};
    for (int k = 0; k < D_; k += 32) {
        v16h a[2], b[4];
#pragma unroll
        for (int i = 0; i < 2; ++i) a[i] = load_frag(A0 + (size_t)i * 16 * D_ + k, D_);
#pragma unroll
        for (int j = 0; j < 4; ++j) b[j] = load_frag(B0 + (size_t)j * 16 * D_ + k, D_);
#pragma unroll
        for (int i = 0; i < 2; ++i)
#pragma unroll
            for (int j = 0; j < 4; ++j)
                acc[i][j] = wmma_f16(a[i], b[j], acc[i][j]);
    }
    int lane = threadIdx.x & 31;
    int cn = lane & 15, cm0 = (lane >> 4) << 3;
    int n0  = tn * 64;
    int sec = n0 >> 10;                           // 0=q 1=k 2=v (64 | 1024)
    int h   = (n0 & (D_ - 1)) >> 6;               // head constant per macro-tile
#pragma unroll
    for (int j = 0; j < 4; ++j) {
        int hd = j * 16 + cn;                     // n0 is 64-aligned
        float bv = bias[n0 + j * 16 + cn];
#pragma unroll
        for (int i = 0; i < 2; ++i)
#pragma unroll
            for (int g = 0; g < 8; ++g) {
                int m = tm * 32 + i * 16 + cm0 + g;
                int b_ = m >> 11, t = m & (T_ - 1);
                size_t bh = (size_t)(b_ * H_ + h);
                float v = acc[i][j][g] + bv;
                if (sec == 0)      Q [(bh * T_ + t) * HD_ + hd] = (_Float16)(v * 0.125f);
                else if (sec == 1) Kb[(bh * T_ + t) * HD_ + hd] = (_Float16)v;
                else               Vt[(bh * HD_ + hd) * T_ + t] = (_Float16)v;
            }
    }
}

// ---------------------------------------------------------- flash attention --
// grid = (B*H, T/64), block = 128 (4 waves).  The 4 waves share one (b,h) and
// a 64-query span; K/V tiles for each 32-key step are staged once per block
// into LDS by the Tensor Data Mover (wave 0 issues, s_wait_tensorcnt, barrier),
// then all waves consume them as ds_load_b128 WMMA fragments.
__global__ void __launch_bounds__(128, 1)
flash_attn(const _Float16* __restrict__ Q,
           const _Float16* __restrict__ Kb,
           const _Float16* __restrict__ Vt,
           _Float16* __restrict__ Oh) {
    __shared__ _Float16 Klds[32 * 64];            // [key][feature]   4 KB
    __shared__ _Float16 Vlds[64 * 32];            // [feature][key]   4 KB
    __shared__ _Float16 Pst[4][16 * 32];          // per-wave P stage 4 KB
    int wave = threadIdx.x >> 5;
    int lane = threadIdx.x & 31;
    int bh   = blockIdx.x;                        // 0..31
    int qt   = blockIdx.y * 4 + wave;             // 0..127
    int b    = bh >> 4, h = bh & (H_ - 1);
    int q0   = qt * 16;

    const _Float16* Qp = Q  + ((size_t)bh * T_ + q0) * HD_;
    const _Float16* Kp = Kb +  (size_t)bh * T_ * HD_;
    const _Float16* Vp = Vt +  (size_t)bh * HD_ * T_;

    v16h aq0 = load_frag(Qp,      HD_);           // features 0..31
    v16h aq1 = load_frag(Qp + 32, HD_);           // features 32..63

    v8f acc[4] = {};
    float mrow[8], lrow[8];
    for (int g = 0; g < 8; ++g) { mrow[g] = -1e30f; lrow[g] = 0.0f; }

    int cn = lane & 15, cm0 = (lane >> 4) << 3;
    _Float16* P = &Pst[wave][0];
    int nkt = blockIdx.y * 2 + 2;                 // causal: key tiles for 64-q span

    for (int kt = 0; kt < nkt; ++kt) {
        int ks = kt * 32;
        // ---- stage K (32x64, contiguous) and V (64x32, stride T) tiles ----
#if HAVE_TDM
        if (wave == 0) {
            tdm_load_2d(lds_off(Klds), Kp + (size_t)ks * HD_, 64, 32, 64);
            tdm_load_2d(lds_off(Vlds), Vp + ks,               32, 64, T_);
            asm volatile("s_wait_tensorcnt 0x0" ::: "memory");
        }
#else
        for (int i = threadIdx.x; i < (32 * 64) / 8; i += blockDim.x)
            ((v8h*)Klds)[i] = ((const v8h*)(Kp + (size_t)ks * HD_))[i];
        for (int i = threadIdx.x; i < (64 * 32) / 8; i += blockDim.x) {
            int f = (i * 8) >> 5, kk = (i * 8) & 31;
            ((v8h*)Vlds)[i] = *(const v8h*)(Vp + (size_t)f * T_ + ks + kk);
        }
#endif
        __syncthreads();

        // ---- S = Q @ K^T for 32 keys (two 16-col tiles, K-depth 64) ----
        v8f s0 = {}, s1 = {};
        s0 = wmma_f16(aq0, load_frag(Klds,            64), s0);
        s0 = wmma_f16(aq1, load_frag(Klds + 32,       64), s0);
        s1 = wmma_f16(aq0, load_frag(Klds + 16 * 64,      64), s1);
        s1 = wmma_f16(aq1, load_frag(Klds + 16 * 64 + 32, 64), s1);

        // ---- causal mask + online softmax (rows span 16 lanes / VGPR) ----
        float corr[8];
#pragma unroll
        for (int g = 0; g < 8; ++g) {
            int q = q0 + cm0 + g;
            float a = s0[g]; if (ks + cn      > q) a = -1e30f;
            float c = s1[g]; if (ks + 16 + cn > q) c = -1e30f;
            float t = fmaxf(a, c);
            t = fmaxf(t, __shfl_xor(t, 1));
            t = fmaxf(t, __shfl_xor(t, 2));
            t = fmaxf(t, __shfl_xor(t, 4));
            t = fmaxf(t, __shfl_xor(t, 8));
            float mn = fmaxf(mrow[g], t);
            float p0 = __expf(a - mn);
            float p1 = __expf(c - mn);
            s0[g] = p0; s1[g] = p1;
            float rs = p0 + p1;
            rs += __shfl_xor(rs, 1);
            rs += __shfl_xor(rs, 2);
            rs += __shfl_xor(rs, 4);
            rs += __shfl_xor(rs, 8);
            float cr = __expf(mrow[g] - mn);
            lrow[g] = lrow[g] * cr + rs;
            mrow[g] = mn;
            corr[g] = cr;
        }
#pragma unroll
        for (int nc = 0; nc < 4; ++nc)
#pragma unroll
            for (int g = 0; g < 8; ++g)
                acc[nc][g] *= corr[g];

        // ---- relayout P: C-layout f32 -> LDS -> A-layout f16 ----
#pragma unroll
        for (int g = 0; g < 8; ++g) {
            int r = cm0 + g;
            P[r * 32 + cn]      = (_Float16)s0[g];
            P[r * 32 + 16 + cn] = (_Float16)s1[g];
        }
        asm volatile("s_wait_dscnt 0x0" ::: "memory");
        v16h ap = load_frag(P, 32);

        // ---- O += P @ V  (V tile in LDS is [feature][key]) ----
#pragma unroll
        for (int nc = 0; nc < 4; ++nc) {
            v16h bv = load_frag(Vlds + (nc * 16) * 32, 32);
            acc[nc] = wmma_f16(ap, bv, acc[nc]);
        }
        __syncthreads();                           // before next tile overwrite
    }

    // ---- epilogue: normalize, scatter to Oh[B*T, D] (f16) ----
#pragma unroll
    for (int nc = 0; nc < 4; ++nc)
#pragma unroll
        for (int g = 0; g < 8; ++g) {
            int q = q0 + cm0 + g;
            size_t idx = ((size_t)(b * T_ + q)) * D_ + h * HD_ + nc * 16 + cn;
            Oh[idx] = (_Float16)(acc[nc][g] / lrow[g]);
        }
}

// --------------------------------------------------------------- out GEMM --
// out[4096,1024] = Oh[4096,1024] @ WoutT'(col-major f16) + bout   (fp32 out)
// Same 32x64 register-blocked scheme.
__global__ void __launch_bounds__(256, 1)
gemm_out(const _Float16* __restrict__ Oh,
         const _Float16* __restrict__ WT,
         const float* __restrict__ bias,
         float* __restrict__ out) {
    int wg = (blockIdx.x * blockDim.x + threadIdx.x) >> 5;
    const int TN = D_ / 64;                        // 16
    int tm = wg / TN, tn = wg % TN;
    const _Float16* A0 = Oh + (size_t)tm * 32 * D_;
    const _Float16* B0 = WT + (size_t)tn * 64 * D_;
    v8f acc[2][4] = {};
    for (int k = 0; k < D_; k += 32) {
        v16h a[2], b[4];
#pragma unroll
        for (int i = 0; i < 2; ++i) a[i] = load_frag(A0 + (size_t)i * 16 * D_ + k, D_);
#pragma unroll
        for (int j = 0; j < 4; ++j) b[j] = load_frag(B0 + (size_t)j * 16 * D_ + k, D_);
#pragma unroll
        for (int i = 0; i < 2; ++i)
#pragma unroll
            for (int j = 0; j < 4; ++j)
                acc[i][j] = wmma_f16(a[i], b[j], acc[i][j]);
    }
    int lane = threadIdx.x & 31;
    int cn = lane & 15, cm0 = (lane >> 4) << 3;
#pragma unroll
    for (int j = 0; j < 4; ++j) {
        int n = tn * 64 + j * 16 + cn;
        float bv = bias[n];
#pragma unroll
        for (int i = 0; i < 2; ++i)
#pragma unroll
            for (int g = 0; g < 8; ++g) {
                int m = tm * 32 + i * 16 + cm0 + g;
                out[(size_t)m * D_ + n] = acc[i][j][g] + bv;
            }
    }
}

// -------------------------------------------------------------------------
extern "C" void kernel_launch(void* const* d_in, const int* in_sizes, int n_in,
                              void* d_out, int out_size, void* d_ws, size_t ws_size,
                              hipStream_t stream) {
    const float* x    = (const float*)d_in[0];
    const float* Wqkv = (const float*)d_in[1];
    const float* bqkv = (const float*)d_in[2];
    const float* Wout = (const float*)d_in[3];
    const float* bout = (const float*)d_in[4];
    float* out = (float*)d_out;

    char* ws = (char*)d_ws;
    const size_t NTOK = (size_t)B_ * T_;                 // 4096
    _Float16* xh    = (_Float16*)(ws);                   //  8 MB
    _Float16* WqkvT = (_Float16*)(ws + (8u  << 20));     //  6 MB
    _Float16* WoutT = (_Float16*)(ws + (14u << 20));     //  2 MB
    _Float16* Qb    = (_Float16*)(ws + (16u << 20));     //  8 MB
    _Float16* Kb    = (_Float16*)(ws + (24u << 20));     //  8 MB
    _Float16* Vt    = (_Float16*)(ws + (32u << 20));     //  8 MB
    _Float16* Oh    = (_Float16*)(ws + (40u << 20));     //  8 MB  (48 MB total)

    cvt_f32_to_f16<<<4096, 256, 0, stream>>>(x, xh, (int)(NTOK * D_));
    transpose_cvt<<<dim3((3 * D_) / 32, D_ / 32), dim3(32, 8), 0, stream>>>(
        Wqkv, WqkvT, D_, 3 * D_);
    transpose_cvt<<<dim3(D_ / 32, D_ / 32), dim3(32, 8), 0, stream>>>(
        Wout, WoutT, D_, D_);

    // (4096/32)*(3072/64) = 6144 waves, 8 waves / 256-thread block
    gemm_qkv<<<768, 256, 0, stream>>>(xh, WqkvT, bqkv, Qb, Kb, Vt);

    flash_attn<<<dim3(B_ * H_, T_ / 64), 128, 0, stream>>>(Qb, Kb, Vt, Oh);

    // (4096/32)*(1024/64) = 2048 waves
    gemm_out<<<256, 256, 0, stream>>>(Oh, WoutT, bout, out);
}
